// GATv2_66589172957299
// MI455X (gfx1250) — matrix operand order repaired and checked by reading, not compiled, not run on hardware
//
#include <hip/hip_runtime.h>
#include <hip/hip_bf16.h>
#include <math.h>

#define NN      50000
#define NE      800000
#define IN_DIM  256
#define HID     64
#define HEADS0  4
#define F0      256      // HEADS0*HID
#define NC      64       // NUM_CLASSES
#define SLOPE   0.2f

typedef __attribute__((ext_vector_type(16))) __bf16 v16bf;
typedef __attribute__((ext_vector_type(8)))  __bf16 v8bf;
typedef __attribute__((ext_vector_type(8)))  float  v8f;

// ---------- monotonic float<->uint key for atomicMax-based segment max ----------
__device__ __forceinline__ unsigned fkey(float f) {
    unsigned u = __float_as_uint(f);
    return (u & 0x80000000u) ? ~u : (u | 0x80000000u);
}
__device__ __forceinline__ float funkey(unsigned k) {
    unsigned u = (k & 0x80000000u) ? (k ^ 0x80000000u) : ~k;
    return __uint_as_float(u);
}
__device__ __forceinline__ float lrelu(float v) { return v > 0.f ? v : SLOPE * v; }

// ---------- init ----------
__global__ void fill_zero_u32(unsigned* __restrict__ p, long long n) {
    long long i = (long long)blockIdx.x * blockDim.x + threadIdx.x;
    if (i < n) p[i] = 0u;
}

// ---------- fp32 -> bf16 conversions ----------
__global__ void conv_bf16(const float* __restrict__ src, __bf16* __restrict__ dst, long long n) {
    long long i = (long long)blockIdx.x * blockDim.x + threadIdx.x;
    if (i < n) dst[i] = (__bf16)src[i];
}
// W [K, Nc] fp32 row-major -> Wt [Nc, K] bf16 (K = 256 fixed)
__global__ void conv_wt(const float* __restrict__ W, __bf16* __restrict__ Wt, int Nc) {
    int t = blockIdx.x * blockDim.x + threadIdx.x;
    if (t >= 256 * Nc) return;
    int n = t % Nc, k = t / Nc;
    Wt[(size_t)n * 256 + k] = (__bf16)W[(size_t)k * Nc + n];
}

// ---------- WMMA GEMM: C[M,Nc] = A[M,256] * B[256,Nc], B given transposed [Nc,256] ----------
// one wave -> 16 rows x 64 cols (4 wmma accumulators), K = 256
__global__ void __launch_bounds__(128)
wmma_gemm_bf16(const __bf16* __restrict__ A, const __bf16* __restrict__ Bt,
               float* __restrict__ C, int M_tiles, int N_groups, int Nc) {
    const int lane = threadIdx.x & 31;
    const int wave = blockIdx.x * (blockDim.x >> 5) + (threadIdx.x >> 5);
    if (wave >= M_tiles * N_groups) return;          // wave-uniform exit
    const int mt = wave / N_groups;
    const int ng = wave - mt * N_groups;
    const int m0 = mt * 16;
    const int n0 = ng * 64;
    const int l15    = lane & 15;
    const int hiHalf = (lane >> 4) & 1;
    const int koffA  = hiHalf ? 16 : 0;              // A: run  [k0+off,+16)
    const int koffB  = hiHalf ? 16 : 0;              // B: run  [k0+off,+16)

    v8f acc[4];
#pragma unroll
    for (int t = 0; t < 4; ++t)
#pragma unroll
        for (int i = 0; i < 8; ++i) acc[t][i] = 0.f;

    const __bf16* Abase = A + (size_t)(m0 + l15) * 256 + koffA;
#pragma unroll 2
    for (int k0 = 0; k0 < 256; k0 += 32) {
        v8bf alo = *(const v8bf*)(Abase + k0);
        v8bf ahi = *(const v8bf*)(Abase + k0 + 8);
        v16bf a;
#pragma unroll
        for (int i = 0; i < 8; ++i) { a[i] = alo[i]; a[i + 8] = ahi[i]; }
#pragma unroll
        for (int t = 0; t < 4; ++t) {
            const __bf16* bp = Bt + (size_t)(n0 + t * 16 + l15) * 256 + k0 + koffB;
            v16bf b = *(const v16bf*)bp;             // 32B contiguous
            acc[t] = __builtin_amdgcn_wmma_f32_16x16x32_bf16(
                false, a, false, b, (short)0, acc[t], false, false);
        }
    }
    const int crow0 = m0 + hiHalf * 8;
#pragma unroll
    for (int t = 0; t < 4; ++t) {
        float* cp = C + (size_t)crow0 * Nc + n0 + t * 16 + l15;
#pragma unroll
        for (int r = 0; r < 8; ++r) cp[(size_t)r * Nc] = acc[t][r];
    }
}

// ======================= layer 0 edge kernels (H=4, D=64, F=256) =======================
// wave per edge; lane handles 8 contiguous floats (head = lane/8)
__global__ void __launch_bounds__(256)
edge_scores_h4(const float* __restrict__ feat, const int* __restrict__ src,
               const int* __restrict__ dst, const float* __restrict__ attn,
               float* __restrict__ ebuf, unsigned* __restrict__ mkey) {
    int gid = blockIdx.x * blockDim.x + threadIdx.x;
    int e = gid >> 5, lane = gid & 31;
    if (e >= NE) return;
    int s = src[e], d = dst[e];
    const float4* fs = (const float4*)(feat + (size_t)s * F0);
    const float4* fd = (const float4*)(feat + (size_t)d * F0);
    const float4* aw = (const float4*)attn;          // attn flat [H*D]=256
    float part = 0.f;
#pragma unroll
    for (int q = 0; q < 2; ++q) {
        float4 a = fs[lane * 2 + q], b = fd[lane * 2 + q], w = aw[lane * 2 + q];
        part += lrelu(a.x + b.x) * w.x + lrelu(a.y + b.y) * w.y
              + lrelu(a.z + b.z) * w.z + lrelu(a.w + b.w) * w.w;
    }
    part += __shfl_xor(part, 1);
    part += __shfl_xor(part, 2);
    part += __shfl_xor(part, 4);                     // sum over 8-lane head group
    int h = lane >> 3;
    if ((lane & 7) == 0) {
        ebuf[(size_t)e * 4 + h] = part;
        atomicMax(&mkey[(size_t)d * 4 + h], fkey(part));
    }
}

__global__ void __launch_bounds__(256)
edge_exp_h4(float* __restrict__ ebuf, const int* __restrict__ dst,
            const unsigned* __restrict__ mkey, float* __restrict__ ssum) {
    long long gid = (long long)blockIdx.x * blockDim.x + threadIdx.x;
    if (gid >= (long long)NE * 4) return;
    int e = (int)(gid >> 2), h = (int)(gid & 3);
    int d = dst[e];
    float m  = funkey(mkey[(size_t)d * 4 + h]);
    float ex = expf(ebuf[gid] - m);
    ebuf[gid] = ex;
    atomicAdd(&ssum[(size_t)d * 4 + h], ex);
}

__global__ void __launch_bounds__(256)
edge_agg_h4(const float* __restrict__ feat, const int* __restrict__ src,
            const int* __restrict__ dst, const float* __restrict__ ebuf,
            const float* __restrict__ ssum, float* __restrict__ out) {
    int gid = blockIdx.x * blockDim.x + threadIdx.x;
    int e = gid >> 5, lane = gid & 31;
    if (e >= NE) return;
    int s = src[e], d = dst[e];
    int h = lane >> 3;
    float alpha = ebuf[(size_t)e * 4 + h] / ssum[(size_t)d * 4 + h];
    const float4* fs = (const float4*)(feat + (size_t)s * F0);
    float* op = out + (size_t)d * F0 + lane * 8;
#pragma unroll
    for (int q = 0; q < 2; ++q) {
        float4 a = fs[lane * 2 + q];
        atomicAdd(op + q * 4 + 0, alpha * a.x);
        atomicAdd(op + q * 4 + 1, alpha * a.y);
        atomicAdd(op + q * 4 + 2, alpha * a.z);
        atomicAdd(op + q * 4 + 3, alpha * a.w);
    }
}

// ELU in place + bf16 copy for next GEMM
__global__ void elu_store(float* __restrict__ h, __bf16* __restrict__ hb, long long n) {
    long long i = (long long)blockIdx.x * blockDim.x + threadIdx.x;
    if (i >= n) return;
    float v = h[i];
    float r = v > 0.f ? v : (expf(v) - 1.f);
    h[i] = r;
    hb[i] = (__bf16)r;
}

// ======================= layer 1 edge kernels (H=1, D=64) =======================
__global__ void __launch_bounds__(256)
edge_scores_h1(const float* __restrict__ feat, const int* __restrict__ src,
               const int* __restrict__ dst, const float* __restrict__ attn,
               float* __restrict__ ebuf, unsigned* __restrict__ mkey) {
    int gid = blockIdx.x * blockDim.x + threadIdx.x;
    int e = gid >> 5, lane = gid & 31;
    if (e >= NE) return;
    int s = src[e], d = dst[e];
    float2 a = ((const float2*)(feat + (size_t)s * NC))[lane];
    float2 b = ((const float2*)(feat + (size_t)d * NC))[lane];
    float2 w = ((const float2*)attn)[lane];
    float part = lrelu(a.x + b.x) * w.x + lrelu(a.y + b.y) * w.y;
    part += __shfl_xor(part, 1);
    part += __shfl_xor(part, 2);
    part += __shfl_xor(part, 4);
    part += __shfl_xor(part, 8);
    part += __shfl_xor(part, 16);
    if (lane == 0) {
        ebuf[e] = part;
        atomicMax(&mkey[d], fkey(part));
    }
}

__global__ void __launch_bounds__(256)
edge_exp_h1(float* __restrict__ ebuf, const int* __restrict__ dst,
            const unsigned* __restrict__ mkey, float* __restrict__ ssum) {
    int e = blockIdx.x * blockDim.x + threadIdx.x;
    if (e >= NE) return;
    int d = dst[e];
    float ex = expf(ebuf[e] - funkey(mkey[d]));
    ebuf[e] = ex;
    atomicAdd(&ssum[d], ex);
}

__global__ void __launch_bounds__(256)
edge_agg_h1(const float* __restrict__ feat, const int* __restrict__ src,
            const int* __restrict__ dst, const float* __restrict__ ebuf,
            const float* __restrict__ ssum, float* __restrict__ out) {
    int gid = blockIdx.x * blockDim.x + threadIdx.x;
    int e = gid >> 5, lane = gid & 31;
    if (e >= NE) return;
    int s = src[e], d = dst[e];
    float alpha = ebuf[e] / ssum[d];
    float2 a = ((const float2*)(feat + (size_t)s * NC))[lane];
    atomicAdd(out + (size_t)d * NC + lane * 2 + 0, alpha * a.x);
    atomicAdd(out + (size_t)d * NC + lane * 2 + 1, alpha * a.y);
}

// ======================= launch =======================
extern "C" void kernel_launch(void* const* d_in, const int* in_sizes, int n_in,
                              void* d_out, int out_size, void* d_ws, size_t ws_size,
                              hipStream_t stream) {
    const float* x     = (const float*)d_in[0];
    const float* W0    = (const float*)d_in[1];
    const float* attn0 = (const float*)d_in[2];
    const float* W1    = (const float*)d_in[3];
    const float* attn1 = (const float*)d_in[4];
    const int*   src0  = (const int*)d_in[5];
    const int*   dst0  = (const int*)d_in[6];
    const int*   src1  = (const int*)d_in[7];
    const int*   dst1  = (const int*)d_in[8];
    float* out = (float*)d_out;

    // workspace carve-out (region starts 64B-aligned, all but ssum1 256B-aligned)
    char* base = (char*)d_ws;
    __bf16*   xb    = (__bf16*)(base);                            // 25,600,000  (x / h bf16)
    __bf16*   w0t   = (__bf16*)(base + 25600000);                 //    131,072
    __bf16*   w1t   = (__bf16*)(base + 25731072);                 //     32,768
    float*    feat  = (float*)(base + 25763840);                  // 51,200,000  (feat0 / feat1)
    float*    out0  = (float*)(base + 76963840);                  // 51,200,000
    float*    ebuf  = (float*)(base + 128163840);                 // 12,800,000  (e/ex both layers)
    unsigned* mkey0 = (unsigned*)(base + 140963840);              //    800,000
    float*    ssum0 = (float*)(base + 141763840);                 //    800,000
    unsigned* mkey1 = (unsigned*)(base + 142563840);              //    200,000
    float*    ssum1 = (float*)(base + 142763840);                 //    200,000

    const int T = 256;
    auto blocks = [](long long n, int t) { return (unsigned)((n + t - 1) / t); };

    // zero accumulators / keys / output (required every launch)
    fill_zero_u32<<<blocks((long long)NN * F0, T), T, 0, stream>>>((unsigned*)out0, (long long)NN * F0);
    fill_zero_u32<<<blocks((long long)NN * 4, T), T, 0, stream>>>(mkey0, (long long)NN * 4);
    fill_zero_u32<<<blocks((long long)NN * 4, T), T, 0, stream>>>((unsigned*)ssum0, (long long)NN * 4);
    fill_zero_u32<<<blocks(NN, T), T, 0, stream>>>(mkey1, NN);
    fill_zero_u32<<<blocks(NN, T), T, 0, stream>>>((unsigned*)ssum1, NN);
    fill_zero_u32<<<blocks((long long)NN * NC, T), T, 0, stream>>>((unsigned*)out, (long long)NN * NC);

    // ---- layer 0 ----
    conv_bf16<<<blocks((long long)NN * IN_DIM, T), T, 0, stream>>>(x, xb, (long long)NN * IN_DIM);
    conv_wt<<<blocks(256 * 256, T), T, 0, stream>>>(W0, w0t, 256);
    // feat0 = x @ W0 : M_tiles=3125, N_groups=4 (4*64=256 cols)
    wmma_gemm_bf16<<<3125, 128, 0, stream>>>(xb, w0t, feat, 3125, 4, 256);

    edge_scores_h4<<<blocks((long long)NE * 32, T), T, 0, stream>>>(feat, src0, dst0, attn0, ebuf, mkey0);
    edge_exp_h4<<<blocks((long long)NE * 4, T), T, 0, stream>>>(ebuf, dst0, mkey0, ssum0);
    edge_agg_h4<<<blocks((long long)NE * 32, T), T, 0, stream>>>(feat, src0, dst0, ebuf, ssum0, out0);

    // ELU + bf16 (reuse xb as h_bf16)
    elu_store<<<blocks((long long)NN * F0, T), T, 0, stream>>>(out0, xb, (long long)NN * F0);

    // ---- layer 1 ----
    conv_wt<<<blocks(256 * 64, T), T, 0, stream>>>(W1, w1t, 64);
    // feat1 = h @ W1 : M_tiles=3125, N_groups=1 (64 cols); 782*4 waves >= 3125
    wmma_gemm_bf16<<<782, 128, 0, stream>>>(xb, w1t, feat, 3125, 1, 64);

    edge_scores_h1<<<blocks((long long)NE * 32, T), T, 0, stream>>>(feat, src1, dst1, attn1, ebuf, mkey1);
    edge_exp_h1<<<blocks(NE, T), T, 0, stream>>>(ebuf, dst1, mkey1, ssum1);
    edge_agg_h1<<<blocks((long long)NE * 32, T), T, 0, stream>>>(feat, src1, dst1, ebuf, ssum1, out);
}
